// GraphAwareAttention_27410481283592
// MI455X (gfx1250) — compile-verified, matrix operand order checked by reading, not run on hardware
//
#include <hip/hip_runtime.h>
#include <hip/hip_bf16.h>
#include <math.h>

typedef __attribute__((ext_vector_type(16))) _Float16 v16h;
typedef __attribute__((ext_vector_type(8)))  float    v8f;

#define B_T   32
#define VLEN  1024
#define CDIM  128
#define NHEAD 4
#define DH    32

// ---------- WMMA helpers ----------------------------------------------------
__device__ __forceinline__ v8f wmma_f16(v16h a, v16h b, v8f c) {
  // D = A(16x32 f16) * B(32x16 f16) + C(16x16 f32)
  return __builtin_amdgcn_wmma_f32_16x16x32_f16(
      /*neg_a=*/false, a, /*neg_b=*/false, b,
      /*c_mod=*/(short)0, c, /*reuse_a=*/false, /*reuse_b=*/false);
}

// Operand layout, wave32 (A-matrix; identical per-lane pattern works for the
// B-matrix when the source is stored "one N-row per lane"):
// lanes 0-15 : row = lane,     K chunks {0..7} and {16..23}
// lanes 16-31: row = lane-16,  K chunks {8..15} and {24..31}
__device__ __forceinline__ v16h ld_tile_h(const _Float16* p0, int ld, int lane) {
  const _Float16* p = p0 + (lane & 15) * ld + ((lane >> 4) << 3);
  v16h r;
#pragma unroll
  for (int i = 0; i < 8; ++i) r[i] = p[i];
#pragma unroll
  for (int i = 0; i < 8; ++i) r[i + 8] = p[i + 16];
  return r;
}

__device__ __forceinline__ v16h ld_tile_f(const float* p0, int ld, int lane) {
  const float* p = p0 + (lane & 15) * ld + ((lane >> 4) << 3);
  v16h r;
#pragma unroll
  for (int i = 0; i < 8; ++i) r[i] = (_Float16)p[i];
#pragma unroll
  for (int i = 0; i < 8; ++i) r[i + 8] = (_Float16)p[i + 16];
  return r;
}

// ---------- 1) QKV projections: Y = X @ W^T + b  ->  f16, head-split --------
// mode 0: Y[(b*NH+h)*V + v][dd]   (row-major per head, for Q/K)
// mode 1: Y[(b*NH+h)*DH + dd][v]  (transposed per head, for V)
__global__ void proj_kernel(const float* __restrict__ X, const float* __restrict__ W,
                            const float* __restrict__ bias, _Float16* __restrict__ Y,
                            int mode) {
  const int lane = threadIdx.x & 31;
  const int wave = threadIdx.x >> 5;
  const int tile = blockIdx.x * 4 + wave;        // 2048 m-tiles x 8 n-tiles
  const int mt = tile >> 3, nt = tile & 7;

  const float* Arow = X + (long)mt * 16 * CDIM;
  const float* Brow = W + (long)nt * 16 * CDIM;  // B[k][n] = W[n][k]

  v8f c = {};
#pragma unroll
  for (int ks = 0; ks < CDIM / 32; ++ks) {
    v16h a = ld_tile_f(Arow + ks * 32, CDIM, lane);
    v16h b = ld_tile_f(Brow + ks * 32, CDIM, lane);
    c = wmma_f16(a, b, c);
  }

  const int n  = nt * 16 + (lane & 15);
  const float bn = bias[n];
  const int h = n >> 5, dd = n & 31;
#pragma unroll
  for (int r = 0; r < 8; ++r) {
    const int m  = mt * 16 + ((lane >> 4) << 3) + r;
    const int b_ = m >> 10, v_ = m & 1023;
    const float val = c[r] + bn;
    long idx;
    if (mode == 0) idx = ((long)(b_ * NHEAD + h) * VLEN + v_) * DH + dd;
    else           idx = ((long)(b_ * NHEAD + h) * DH + dd) * VLEN + v_;
    Y[idx] = (_Float16)val;
  }
}

// ---------- 2) fused scores + softmax + P@V ---------------------------------
// One workgroup (8 waves) per (bh, q-tile of 16 rows).
//   phase 1: S_lds[16][1024] = scale*Q@K^T + alpha*graph_bias   (64 WMMAs)
//   phase 2: row softmax; write normalized f32 to Pglob (mandatory output),
//            write f16 copy in place into LDS (A-operand for phase 3)
//   phase 3: O_tile(16x32) = P_strip @ V_h, ds_add_f32 reduction (64 WMMAs)
__global__ void attn_fused_kernel(const _Float16* __restrict__ Qh,
                                  const _Float16* __restrict__ Kh,
                                  const _Float16* __restrict__ Vt,
                                  const float* __restrict__ gb,
                                  const float* __restrict__ alpha_p,
                                  float* __restrict__ Pglob,
                                  _Float16* __restrict__ O) {
  extern __shared__ char smem[];
  float* Sl   = (float*)smem;                   // [16][1024] score strip (64 KB)
  float* Oacc = (float*)(smem + 16 * VLEN * 4); // [2][16][16] accumulators (2 KB)

  const int lane = threadIdx.x & 31;
  const int wave = threadIdx.x >> 5;            // 0..7
  const int qt   = blockIdx.x;                  // 0..63
  const int bh   = blockIdx.y;                  // 0..127
  const int lrow = ((lane >> 4) << 3);          // C/D local row base for this lane

  // zero O accumulators (visible to phase 3 via the phase-1/2 barriers)
  Oacc[threadIdx.x]       = 0.f;
  Oacc[threadIdx.x + 256] = 0.f;

  // ----- phase 1: scores into LDS -----
  const _Float16* Qbase = Qh + ((long)bh * VLEN + qt * 16) * DH;
  v16h aq = ld_tile_h(Qbase, DH, lane);
  const float alpha = alpha_p[0];
  const float scale = 0.17677669529663689f;     // dh^-0.5, dh = 32

#pragma unroll
  for (int i = 0; i < 8; ++i) {
    const int kt = wave * 8 + i;
    const _Float16* Kbase = Kh + ((long)bh * VLEN + kt * 16) * DH;
    v16h bkk = ld_tile_h(Kbase, DH, lane);
    v8f c = {};
    c = wmma_f16(aq, bkk, c);
    const int n = kt * 16 + (lane & 15);
#pragma unroll
    for (int r = 0; r < 8; ++r) {
      const int q = qt * 16 + lrow + r;
      Sl[(lrow + r) * VLEN + n] = c[r] * scale + alpha * gb[(long)q * VLEN + n];
    }
  }
  __syncthreads();

  // ----- phase 2: softmax, write f32 probs to global, f16 copy in place -----
#pragma unroll
  for (int rr = 0; rr < 2; ++rr) {
    const int row = wave * 2 + rr;
    float* Srow = Sl + row * VLEN;
    float e[32];
    float m = -3.4e38f;
#pragma unroll
    for (int i = 0; i < 32; ++i) { e[i] = Srow[i * 32 + lane]; m = fmaxf(m, e[i]); }
#pragma unroll
    for (int off = 16; off > 0; off >>= 1) m = fmaxf(m, __shfl_xor(m, off, 32));
    float s = 0.f;
#pragma unroll
    for (int i = 0; i < 32; ++i) { e[i] = __expf(e[i] - m); s += e[i]; }
#pragma unroll
    for (int off = 16; off > 0; off >>= 1) s += __shfl_xor(s, off, 32);
    const float inv = 1.0f / s;

    float* gp = Pglob + ((long)bh * VLEN + qt * 16 + row) * VLEN;
    _Float16* ph = (_Float16*)Srow;             // in-place f16 row (stride 2048 halves)
#pragma unroll
    for (int i = 0; i < 32; ++i) {
      const float p = e[i] * inv;
      gp[i * 32 + lane] = p;                    // mandatory attn_prob output
      ph[i * 32 + lane] = (_Float16)p;          // A-operand for phase 3
    }
  }
  __syncthreads();

  // ----- phase 3: O_tile = P_strip @ V_h -----
  const int nt = wave & 1;                      // d-tile (0,1)
  const int kg = wave >> 1;                     // K-group (0..3), 8 k-steps each
  const _Float16* Pa = (const _Float16*)Sl;     // [16][2048] halves
  const _Float16* Br = Vt + ((long)bh * DH + nt * 16) * VLEN;

  v8f c = {};
#pragma unroll
  for (int i = 0; i < 8; ++i) {
    const int ks = kg * 8 + i;
    v16h a = ld_tile_h(Pa + ks * 32, 2 * VLEN, lane);
    v16h b = ld_tile_h(Br + ks * 32, VLEN, lane);
    c = wmma_f16(a, b, c);
  }
  {
    float* acc = Oacc + nt * 256 + lrow * 16 + (lane & 15);
#pragma unroll
    for (int r = 0; r < 8; ++r) atomicAdd(acc + r * 16, c[r]);   // ds_add_f32
  }
  __syncthreads();

  // write O tile (16 rows x 32 dims) as f16, merged-head layout (B*V, C)
  const int b_ = bh >> 2, h = bh & 3;
#pragma unroll
  for (int t = 0; t < 2; ++t) {
    const int e2 = threadIdx.x + t * 256;       // 0..511 -> [nt][m][n]
    const int n2 = e2 & 15, m2 = (e2 >> 4) & 15, nt2 = e2 >> 8;
    O[((long)(b_ * VLEN + qt * 16 + m2)) * CDIM + h * DH + nt2 * 16 + n2] =
        (_Float16)Oacc[e2];
  }
}

// ---------- 3) out = O @ Wo^T + bo  -> f32 ----------------------------------
__global__ void outproj_kernel(const _Float16* __restrict__ O, const float* __restrict__ Wo,
                               const float* __restrict__ bo, float* __restrict__ Y) {
  const int lane = threadIdx.x & 31;
  const int wave = threadIdx.x >> 5;
  const int tile = blockIdx.x * 4 + wave;        // 2048 m-tiles x 8 n-tiles
  const int mt = tile >> 3, nt = tile & 7;

  const _Float16* Ar = O  + (long)mt * 16 * CDIM;
  const float*    Br = Wo + (long)nt * 16 * CDIM;

  v8f c = {};
#pragma unroll
  for (int ks = 0; ks < CDIM / 32; ++ks) {
    v16h a = ld_tile_h(Ar + ks * 32, CDIM, lane);
    v16h b = ld_tile_f(Br + ks * 32, CDIM, lane);
    c = wmma_f16(a, b, c);
  }

  const int n  = nt * 16 + (lane & 15);
  const float bn = bo[n];
#pragma unroll
  for (int r = 0; r < 8; ++r) {
    const int m = mt * 16 + ((lane >> 4) << 3) + r;
    Y[(long)m * CDIM + n] = c[r] + bn;
  }
}

// ---------------------------------------------------------------------------
extern "C" void kernel_launch(void* const* d_in, const int* in_sizes, int n_in,
                              void* d_out, int out_size, void* d_ws, size_t ws_size,
                              hipStream_t stream) {
  const float* q     = (const float*)d_in[0];
  const float* k     = (const float*)d_in[1];
  const float* v     = (const float*)d_in[2];
  const float* gb    = (const float*)d_in[3];
  const float* Wq    = (const float*)d_in[4];
  const float* bq    = (const float*)d_in[5];
  const float* Wk    = (const float*)d_in[6];
  const float* bk    = (const float*)d_in[7];
  const float* Wv    = (const float*)d_in[8];
  const float* bv    = (const float*)d_in[9];
  const float* Wo    = (const float*)d_in[10];
  const float* bo    = (const float*)d_in[11];
  const float* alpha = (const float*)d_in[12];

  float* out  = (float*)d_out;                              // (B*V, C)
  float* attn = out + (size_t)B_T * VLEN * CDIM;            // (B, h, V, V)

  const size_t HE = (size_t)B_T * NHEAD * VLEN * DH;        // 4,194,304 halves each
  _Float16* Qh = (_Float16*)d_ws;
  _Float16* Kh = Qh + HE;
  _Float16* Vt = Kh + HE;
  _Float16* O  = Vt + HE;                                   // 32 MB f16 workspace

  const dim3 blk(128);
  proj_kernel<<<4096, blk, 0, stream>>>(q, Wq, bq, Qh, 0);
  proj_kernel<<<4096, blk, 0, stream>>>(k, Wk, bk, Kh, 0);
  proj_kernel<<<4096, blk, 0, stream>>>(v, Wv, bv, Vt, 1);

  const int lds_bytes = 16 * VLEN * 4 + 2 * 16 * 16 * 4;    // 67,584 B < 320 KB/WGP
  attn_fused_kernel<<<dim3(64, 128), dim3(256), lds_bytes, stream>>>(
      Qh, Kh, Vt, gb, alpha, attn, O);

  outproj_kernel<<<4096, blk, 0, stream>>>(O, Wo, bo, out);
}
